// Head_46411416600782
// MI455X (gfx1250) — matrix operand order repaired and checked by reading, not compiled, not run on hardware
//
#include <hip/hip_runtime.h>

// Problem constants (match reference)
constexpr int Bc = 4, Tc = 4096, Cc = 512, Hc = 64;

typedef _Float16 v8h  __attribute__((ext_vector_type(8)));
typedef _Float16 v16h __attribute__((ext_vector_type(16)));
typedef float    v8f  __attribute__((ext_vector_type(8)));

union V16H { v16h v; v8h h8[2]; unsigned u[8]; };
union H2   { _Float16 h[2]; unsigned u; };

__device__ __forceinline__ v8f wmma16x16x32(v16h a, v16h b, v8f c) {
  // D = A(16x32 f16) x B(32x16 f16) + C(16x16 f32)
  return __builtin_amdgcn_wmma_f32_16x16x32_f16(false, a, false, b, (short)0, c,
                                                false, false);
}

// ---------------------------------------------------------------------------
// Kernel 1: transpose+convert weights to f16:  Wt[mat][h][c] = W[mat][c][h]
// mat: 0 = Wq, 1 = Wk, 2 = Wv
// ---------------------------------------------------------------------------
__global__ void wconvert_kernel(const float* __restrict__ Wq,
                                const float* __restrict__ Wk,
                                const float* __restrict__ Wv,
                                _Float16* __restrict__ Wt) {
  const int NW = Cc * Hc;
  int i = blockIdx.x * blockDim.x + threadIdx.x;
  if (i >= 3 * NW) return;
  int mat = i / NW, r = i % NW;
  int h = r / Cc, c = r % Cc;
  const float* W = (mat == 0) ? Wq : (mat == 1) ? Wk : Wv;
  Wt[(size_t)mat * NW + (size_t)h * Cc + c] = (_Float16)W[(size_t)c * Hc + h];
}

// ---------------------------------------------------------------------------
// Kernel 2: projections via WMMA.  One wave = 16 rows x 64 cols of one matrix.
//   mat 0 -> Qf[T,H] (pre-scaled by H^-0.5), mat 1 -> Kf[T,H], mat 2 -> Vt[H,T]
// A-frag (x, f32->f16 on the fly): lane m = row, elems0..7 = c: 16*hf..+7,
//   elems8..15 = c: 16*hf+8..+7.
// B-frag (Wt): lane n = h_local, elems0..15 = c: 16*hf..+15 (contiguous b128s).
// ---------------------------------------------------------------------------
__global__ __launch_bounds__(128)
void proj_kernel(const float* __restrict__ x, const _Float16* __restrict__ Wt,
                 _Float16* __restrict__ Qf, _Float16* __restrict__ Kf,
                 _Float16* __restrict__ Vt) {
  const int lane = threadIdx.x & 31, wid = threadIdx.x >> 5;
  const int n = lane & 15, hf = lane >> 4;
  const int t0 = (blockIdx.x * 4 + wid) * 16;
  const int b = blockIdx.y, mat = blockIdx.z;

  const float*    xrow = x  + ((size_t)b * Tc + t0 + n) * Cc;
  const _Float16* w    = Wt + (size_t)mat * Hc * Cc;

  const v8f zero = {0.f,0.f,0.f,0.f,0.f,0.f,0.f,0.f};
  v8f acc[4] = {zero, zero, zero, zero};

  for (int cb = 0; cb < Cc; cb += 32) {
    V16H a;
#pragma unroll
    for (int j = 0; j < 8; ++j) {
      a.v[j]     = (_Float16)xrow[cb + 16 * hf + j];
      a.v[8 + j] = (_Float16)xrow[cb + 16 * hf + 8 + j];
    }
#pragma unroll
    for (int ht = 0; ht < 4; ++ht) {
      const _Float16* wrow = w + (size_t)(ht * 16 + n) * Cc + cb + 16 * hf;
      V16H bw;
      bw.h8[0] = *(const v8h*)(wrow);
      bw.h8[1] = *(const v8h*)(wrow + 8);
      acc[ht] = wmma16x16x32(a.v, bw.v, acc[ht]);
    }
  }

  // ---- branch-free strided store epilogue ----------------------------------
  // D-frag: lane n = h_local (col = ht*16+n), row = t0 + r + 8*hf.
  //   mat 0/1: out[row*Hc + col]   (row-major [T,H])
  //   mat 2  : out[col*Tc + row]   (transposed [H,T])
  _Float16* obase;
  size_t rowStride, colStride;
  float scale;
  if (mat == 0)      { obase = Qf + (size_t)b * Tc * Hc; rowStride = Hc; colStride = 1;  scale = 0.125f; }
  else if (mat == 1) { obase = Kf + (size_t)b * Tc * Hc; rowStride = Hc; colStride = 1;  scale = 1.0f;   }
  else               { obase = Vt + (size_t)b * Hc * Tc; rowStride = 1;  colStride = Tc; scale = 1.0f;   }

  _Float16* op0 = obase + (size_t)(t0 + 8 * hf) * rowStride + (size_t)n * colStride;
#pragma unroll
  for (int ht = 0; ht < 4; ++ht) {
#pragma unroll
    for (int r = 0; r < 8; ++r) {
      op0[(size_t)r * rowStride + (size_t)(ht * 16) * colStride] =
          (_Float16)(acc[ht][r] * scale);
    }
  }
}

// ---------------------------------------------------------------------------
// Kernel 3: flash attention with staircase mask collapsed to a prefix bound.
// One wave handles a 16-query tile; key blocks of 32 up to (blk+1)*128.
//   S^T(16k x 16q) = K(16x64) x Q^T(64x16)   [2 chained wmma per 16-key half]
//   O^T(16h x 16q) += V^T(16x32) x P(32x16)  [4 wmma across H=64]
// ---------------------------------------------------------------------------
__global__ __launch_bounds__(128)
void attn_kernel(const _Float16* __restrict__ Qf, const _Float16* __restrict__ Kf,
                 const _Float16* __restrict__ Vt, float* __restrict__ out) {
  const int lane = threadIdx.x & 31, wid = threadIdx.x >> 5;
  const int n = lane & 15, hf = lane >> 4;
  const int qt = blockIdx.x * 4 + wid;       // 0..255
  const int b  = blockIdx.y;
  const int q0 = qt * 16;

  const _Float16* Qb = Qf + (size_t)b * Tc * Hc;
  const _Float16* Kb = Kf + (size_t)b * Tc * Hc;
  const _Float16* Vb = Vt + (size_t)b * Hc * Tc;

  // Q^T B-frags (lane n = query): elems0..7 = h: 8*hf..+7, elems8..15 = h: 16+8*hf..+7 (+32 for frag 1)
  V16H bq0, bq1;
  {
    const _Float16* qrow = Qb + (size_t)(q0 + n) * Hc + 8 * hf;
    bq0.h8[0] = *(const v8h*)(qrow);
    bq0.h8[1] = *(const v8h*)(qrow + 16);
    bq1.h8[0] = *(const v8h*)(qrow + 32);
    bq1.h8[1] = *(const v8h*)(qrow + 48);
  }

  const v8f zero = {0.f,0.f,0.f,0.f,0.f,0.f,0.f,0.f};
  v8f o[4] = {zero, zero, zero, zero};
  float mrun = -1e30f, lrun = 0.f;

  // staircase mask == softmax over keys [0, (qt/8 + 1)*128)
  const int nsteps = (qt / 8 + 1) * 4;       // 32 keys per step
  for (int st = 0; st < nsteps; ++st) {
    const int s0 = st * 32;

    // Prefetch next key block (K rows and V^T rows) into cache while this
    // step's softmax VALU work runs: lowers to global_prefetch_b8.
    if (st + 1 < nsteps) {
      __builtin_prefetch(Kb + (size_t)(s0 + 32 + n) * Hc, 0, 1);
      __builtin_prefetch(Kb + (size_t)(s0 + 48 + n) * Hc, 0, 1);
      __builtin_prefetch(Vb + (size_t)n * Tc + s0 + 32, 0, 1);
      __builtin_prefetch(Vb + (size_t)(32 + n) * Tc + s0 + 32, 0, 1);
    }

    // ---- S^T = K x Q^T ------------------------------------------------
    V16H ak;                                  // K A-frag: lane m = key_local
    v8f clo = zero, chi = zero;               // keys s0..+15 / s0+16..+31
    const _Float16* kr0 = Kb + (size_t)(s0 + n) * Hc + 8 * hf;
    ak.h8[0] = *(const v8h*)(kr0);            // h: 8*hf..+7
    ak.h8[1] = *(const v8h*)(kr0 + 16);       // h: 16+8*hf..+7
    clo = wmma16x16x32(ak.v, bq0.v, clo);
    ak.h8[0] = *(const v8h*)(kr0 + 32);
    ak.h8[1] = *(const v8h*)(kr0 + 48);
    clo = wmma16x16x32(ak.v, bq1.v, clo);
    const _Float16* kr1 = kr0 + 16 * Hc;
    ak.h8[0] = *(const v8h*)(kr1);
    ak.h8[1] = *(const v8h*)(kr1 + 16);
    chi = wmma16x16x32(ak.v, bq0.v, chi);
    ak.h8[0] = *(const v8h*)(kr1 + 32);
    ak.h8[1] = *(const v8h*)(kr1 + 48);
    chi = wmma16x16x32(ak.v, bq1.v, chi);

    // ---- online softmax (lane = query; keys split across half-waves) ----
    float mloc = -1e30f;
#pragma unroll
    for (int r = 0; r < 8; ++r) mloc = fmaxf(mloc, fmaxf(clo[r], chi[r]));
    mloc = fmaxf(mloc, __shfl_xor(mloc, 16, 32));
    const float mnew = fmaxf(mrun, mloc);
    const float corr = __expf(mrun - mnew);
    mrun = mnew;

    float elo[8], ehi[8], ps = 0.f;
#pragma unroll
    for (int r = 0; r < 8; ++r) {
      elo[r] = __expf(clo[r] - mnew);
      ehi[r] = __expf(chi[r] - mnew);
      ps += elo[r] + ehi[r];
    }
    ps += __shfl_xor(ps, 16, 32);
    lrun = lrun * corr + ps;
#pragma unroll
    for (int ht = 0; ht < 4; ++ht)
#pragma unroll
      for (int r = 0; r < 8; ++r) o[ht][r] = o[ht][r] * corr;

    // ---- pack P into B-frag (elems k = 16*hf + 0..15, lane n = query) ----
    // own clo holds keys r+8*hf; partner half-wave supplies the other 8.
    V16H bp;
#pragma unroll
    for (int j = 0; j < 4; ++j) {
      H2 plo, phi;
      plo.h[0] = (_Float16)elo[2*j]; plo.h[1] = (_Float16)elo[2*j+1];
      phi.h[0] = (_Float16)ehi[2*j]; phi.h[1] = (_Float16)ehi[2*j+1];
      const unsigned qlo = (unsigned)__shfl_xor((int)plo.u, 16, 32);
      const unsigned qhi = (unsigned)__shfl_xor((int)phi.u, 16, 32);
      bp.u[j]     = hf ? qhi   : plo.u;   // k = 16*hf + 2j, +1
      bp.u[4 + j] = hf ? phi.u : qlo;     // k = 16*hf + 8 + 2j, +1
    }

    // ---- O^T += V^T x P ----------------------------------------------
#pragma unroll
    for (int ht = 0; ht < 4; ++ht) {
      const _Float16* vr = Vb + (size_t)(ht * 16 + n) * Tc + s0 + 16 * hf;
      V16H av;                              // lane m = h_local
      av.h8[0] = *(const v8h*)(vr);         // s: 16*hf..+7
      av.h8[1] = *(const v8h*)(vr + 8);     // s: 16*hf+8..+7
      o[ht] = wmma16x16x32(av.v, bp.v, o[ht]);
    }
  }

  // ---- epilogue: out[b, q0+n, h] = O^T[h, n] / l --------------------------
  const float inv = 1.f / lrun;
  float* orow = out + ((size_t)b * Tc + q0 + n) * Hc;
#pragma unroll
  for (int ht = 0; ht < 4; ++ht)
#pragma unroll
    for (int r = 0; r < 8; ++r)
      orow[ht * 16 + r + 8 * hf] = o[ht][r] * inv;
}

// ---------------------------------------------------------------------------
extern "C" void kernel_launch(void* const* d_in, const int* in_sizes, int n_in,
                              void* d_out, int out_size, void* d_ws, size_t ws_size,
                              hipStream_t stream) {
  (void)in_sizes; (void)n_in; (void)out_size; (void)ws_size;
  // setup_inputs order: x, Wk, Wq, Wv
  const float* x  = (const float*)d_in[0];
  const float* Wk = (const float*)d_in[1];
  const float* Wq = (const float*)d_in[2];
  const float* Wv = (const float*)d_in[3];

  // workspace carve (halfs): Wt[3][H][C] | Qf[B,T,H] | Kf[B,T,H] | Vt[B,H,T]
  _Float16* Wt = (_Float16*)d_ws;
  _Float16* Qf = Wt + (size_t)3 * Hc * Cc;
  _Float16* Kf = Qf + (size_t)Bc * Tc * Hc;
  _Float16* Vt = Kf + (size_t)Bc * Tc * Hc;
  float* out = (float*)d_out;

  const int NW3 = 3 * Cc * Hc;
  wconvert_kernel<<<dim3((NW3 + 255) / 256), dim3(256), 0, stream>>>(Wq, Wk, Wv, Wt);
  proj_kernel<<<dim3(Tc / 64, Bc, 3), dim3(128), 0, stream>>>(x, Wt, Qf, Kf, Vt);
  attn_kernel<<<dim3(Tc / 64, Bc), dim3(128), 0, stream>>>(Qf, Kf, Vt, out);
}
